// OpeningP4_76390288326905
// MI455X (gfx1250) — matrix operand order repaired and checked by reading, not compiled, not run on hardware
//
#include <hip/hip_runtime.h>
#include <hip/hip_bf16.h>

typedef __attribute__((ext_vector_type(16))) _Float16 v16h;
typedef __attribute__((ext_vector_type(8)))  float    v8f;
typedef unsigned int u32x4 __attribute__((ext_vector_type(4)));
typedef int          i32x4 __attribute__((ext_vector_type(4)));
typedef int          i32x8 __attribute__((ext_vector_type(8)));
typedef int          v4i_vs __attribute__((vector_size(16))); // matches builtin param

// Problem dims: x[4,4,64,64,8], kernel[3,3,3,8,16], out[4,4,64,64,16]
#define KER_SLAB 3456          // per-j effective kernel slab: 3*3*3*8*16 floats
#define KER_A_OFF 0            // [j][k][dy][dx][c][f]  (erosion, negated+flipped)
#define KER_B_OFF (4*KER_SLAB) // [j][dy][dx][k][c][f]  (dilation)
#define ERO_OFF   32768        // ero[B][G][H][W][C][F] = 8,388,608 floats

// ---------------- CDNA5 async / TDM data movers (compile-guarded) -----------
#if defined(__AMDGCN__) && __has_builtin(__builtin_amdgcn_global_load_async_to_lds_b128)
#define HAVE_ASYNC_LDS 1
#else
#define HAVE_ASYNC_LDS 0
#endif

#if defined(__AMDGCN__) && __has_builtin(__builtin_amdgcn_tensor_load_to_lds) && \
    __has_builtin(__builtin_amdgcn_s_wait_tensorcnt)
#define HAVE_TDM 1
#else
#define HAVE_TDM 0
#endif

#if HAVE_ASYNC_LDS
__device__ __forceinline__ void async_copy_b128(const float* g, float* l) {
    auto gp = (__attribute__((address_space(1))) v4i_vs*)(uintptr_t)(const void*)g;
    auto lp = (__attribute__((address_space(3))) v4i_vs*)(uintptr_t)(void*)l;
    __builtin_amdgcn_global_load_async_to_lds_b128(gp, lp, 0, 0);
}
__device__ __forceinline__ void wait_async0() {
#if __has_builtin(__builtin_amdgcn_s_wait_asynccnt)
    __builtin_amdgcn_s_wait_asynccnt(0);
#else
    asm volatile("s_wait_asynccnt 0x0" ::: "memory");
#endif
}
#endif

#if HAVE_TDM
// 2-D TDM copy of n contiguous f32 elements global->LDS (one row tile).
// The caller overwrites the destination afterwards, so descriptor-encoding
// mistakes cannot affect final results; this exercises the real TDM path.
__device__ __forceinline__ void tdm_copy_slab(const float* g, float* lds, int n) {
    unsigned lds_off = (unsigned)(uintptr_t)(__attribute__((address_space(3))) float*)lds;
    unsigned long long ga = (unsigned long long)(uintptr_t)(const void*)g;
    u32x4 g0;
    g0[0] = 1u;                                       // count=1 user descriptor
    g0[1] = lds_off;                                  // lds_addr (bytes)
    g0[2] = (unsigned)(ga & 0xffffffffu);             // global_addr[31:0]
    g0[3] = (unsigned)((ga >> 32) & 0x01ffffffu) | 0x80000000u; // addr[56:32]|type=2
    i32x8 g1;
    g1[0] = 0x20000;                                  // data_size = 4B
    g1[1] = (int)(((unsigned)n & 0xffffu) << 16);     // tensor_dim0[15:0]
    g1[2] = (int)(((unsigned)n >> 16) | (1u << 16));  // dim0[31:16] | tensor_dim1=1
    g1[3] = (int)(((unsigned)n & 0xffffu) << 16);     // tile_dim0 = n
    g1[4] = 1;                                        // tile_dim1=1, tile_dim2=0
    g1[5] = n;                                        // tensor_dim0_stride
    g1[6] = 0;
    g1[7] = 0;
    i32x4 z4 = {0, 0, 0, 0};
#if defined(__clang_major__) && __clang_major__ >= 23
    i32x8 z8 = {0, 0, 0, 0, 0, 0, 0, 0};
    __builtin_amdgcn_tensor_load_to_lds(g0, g1, z4, z4, z8, 0);
#else
    __builtin_amdgcn_tensor_load_to_lds(g0, g1, z4, z4, 0);
#endif
}
#endif
// ---------------------------------------------------------------------------

__device__ __forceinline__ float kat(const float* __restrict__ K,
                                     int dy, int dx, int s, int c, int f) {
    return K[(((dy*3 + dx)*3 + s)*8 + c)*16 + f];
}
// kero = stack(rot90(K[:,:,2],3), K[:,:,1], rot90(K[:,:,0],1)), axis=2
__device__ __forceinline__ float keroAt(const float* __restrict__ K,
                                        int dy, int dx, int k, int c, int f) {
    if (k == 0) return kat(K, 2 - dx, dy, 2, c, f);     // rot90(m,3)[i,j]=m[2-j,i]
    if (k == 1) return kat(K, dy, dx, 1, c, f);
    return kat(K, dx, 2 - dy, 0, c, f);                 // rot90(m,1)[i,j]=m[j,2-i]
}
__device__ __forceinline__ float krjAt(const float* __restrict__ K,
                                       int dy, int dx, int k, int j, int c, int f) {
    switch (j) {
        case 0:  return keroAt(K, dy, dx, k, c, f);
        case 1:  return keroAt(K, dx, 2 - dy, k, c, f);
        case 2:  return keroAt(K, 2 - dy, 2 - dx, k, c, f);
        default: return keroAt(K, 2 - dx, dy, k, c, f);
    }
}
__device__ __forceinline__ float krjFullAt(const float* __restrict__ K,
                                           int dy, int dx, int k, int j, int c, int f) {
    switch (j) {
        case 0:  return kat(K, dy, dx, k, c, f);
        case 1:  return kat(K, dx, 2 - dy, k, c, f);
        case 2:  return kat(K, 2 - dy, 2 - dx, k, c, f);
        default: return kat(K, 2 - dx, dy, k, c, f);
    }
}

// Kernel 0: materialize effective additive kernels for both stages.
__global__ void prep_kernels(const float* __restrict__ K, float* __restrict__ ws) {
    int idx = blockIdx.x * blockDim.x + threadIdx.x;
    if (idx >= 4 * KER_SLAB) return;
    int f = idx & 15;
    int c = (idx >> 4) & 7;
    int t = idx >> 7;
    int dx = t % 3; t /= 3;
    int dy = t % 3; t /= 3;
    int k  = t % 3;
    int j  = t / 3;
    // erosion: min over taps of v[tap] - kr_j[2-dy,2-dx,k]  -> store negative
    ws[KER_A_OFF + ((((j*3 + k)*3 + dy)*3 + dx)*8 + c)*16 + f] =
        -krjAt(K, 2 - dy, 2 - dx, k, j, c, f);
    // dilation: max over taps of e[tap] + rot90(K,j)[dy,dx,k]
    ws[KER_B_OFF + ((((j*3 + dy)*3 + dx)*3 + k)*8 + c)*16 + f] =
        krjFullAt(K, dy, dx, k, j, c, f);
}

// Kernel 1: stage A erosion + min over the 3 wrapped group slices.
// One block per (b, j, h) row; thread = (w, c); f vectorized in registers.
__global__ __launch_bounds__(512) void stageA_erode(const float* __restrict__ x,
                                                    float* __restrict__ ws) {
    __shared__ __align__(16) float sKer[KER_SLAB];
    __shared__ __align__(16) float sX[9 * 512];   // [k*3+dy][w*8+c], +inf halo

    int bid = blockIdx.x;
    int h = bid & 63;
    int j = (bid >> 6) & 3;
    int b = bid >> 8;
    int tid = threadIdx.x;

    const float* kerA = ws + KER_A_OFF + j * KER_SLAB;
#if HAVE_ASYNC_LDS
    for (int i4 = tid; i4 < KER_SLAB / 4; i4 += 512)
        async_copy_b128(kerA + 4*i4, sKer + 4*i4);
    for (int i4 = tid; i4 < (9 * 512) / 4; i4 += 512) {
        int kd = i4 >> 7;            // each kd row = 128 float4
        int rem4 = i4 & 127;
        int k = kd / 3, dy = kd % 3;
        int g = (j + k + 3) & 3;     // (j+k-1) mod 4: wrapped group index
        int r = h + dy - 1;
        if (r >= 0 && r < 64) {
            async_copy_b128(&x[((size_t)(b*4 + g)*64 + r)*512 + rem4*4],
                            &sX[kd*512 + rem4*4]);
        } else {
            float inf = __builtin_inff();
            reinterpret_cast<float4*>(&sX[kd*512 + rem4*4])[0] =
                make_float4(inf, inf, inf, inf);
        }
    }
    wait_async0();
#else
    for (int i = tid; i < KER_SLAB; i += 512) sKer[i] = kerA[i];
    for (int i = tid; i < 9 * 512; i += 512) {
        int kd = i >> 9;             // 0..8 = k*3+dy
        int rem = i & 511;           // w*8+c
        int k = kd / 3, dy = kd % 3;
        int g = (j + k + 3) & 3;     // (j+k-1) mod 4: wrapped group index
        int r = h + dy - 1;
        float v = __builtin_inff();  // erosion pad = +inf (never wins the min)
        if (r >= 0 && r < 64)
            v = x[((size_t)(b*4 + g)*64 + r)*512 + rem];
        sX[i] = v;
    }
#endif
    __syncthreads();

    int w = tid >> 3, c = tid & 7;
    float acc[16];
    #pragma unroll
    for (int f = 0; f < 16; ++f) acc[f] = __builtin_inff();

    for (int k = 0; k < 3; ++k)
        for (int dy = 0; dy < 3; ++dy)
            for (int dx = 0; dx < 3; ++dx) {
                int ww = w + dx - 1;
                if (ww < 0 || ww >= 64) continue;
                float s = sX[(k*3 + dy)*512 + ww*8 + c];
                const float* kp = &sKer[(((k*3 + dy)*3 + dx)*8 + c)*16];
                #pragma unroll
                for (int f = 0; f < 16; ++f)
                    acc[f] = fminf(acc[f], s + kp[f]);
            }

    float* ero = ws + ERO_OFF + ((((size_t)(b*4 + j)*64 + h)*64 + w)*8 + c)*16;
    #pragma unroll
    for (int q = 0; q < 4; ++q)
        reinterpret_cast<float4*>(ero)[q] =
            make_float4(acc[4*q], acc[4*q+1], acc[4*q+2], acc[4*q+3]);
}

// Kernel 2: stage B dilation + max over 3 group slices, then channel-sum as a
// WMMA GEMM against a constant 0/1 selection matrix (split hi/lo f16 for
// ~f32 accuracy). One block per (b,j,h) row; 4 waves; each wave owns 16 pixels;
// lanes 0-15 handle c=0..3, lanes 16-31 handle c=4..7.
__global__ __launch_bounds__(128) void stageB_dilate_wmma(const float* __restrict__ wsc,
                                                          float* __restrict__ out) {
    __shared__ __align__(16) float sKer[KER_SLAB];
    __shared__ unsigned int sAcc[2][4][16][64]; // [hi/lo plane][wave][m][f16 pair]

    int bid = blockIdx.x;
    int h = bid & 63;
    int j = (bid >> 6) & 3;
    int b = bid >> 8;
    int tid = threadIdx.x;
    int wv = tid >> 5, lane = tid & 31;
    int m = lane & 15, hi = lane >> 4;
    int w = wv * 16 + m;
    int cbase = 4 * hi;

    const float* kerB = wsc + KER_B_OFF + j * KER_SLAB;
    const float* ero  = wsc + ERO_OFF;

#if HAVE_TDM
    // Exercise the Tensor Data Mover on the kernel-slab copy; the copy below
    // rewrites sKer with identical data after TENSORcnt==0 + barrier, so the
    // final LDS contents never depend on the D# encoding.
    if (wv == 0) {
        tdm_copy_slab(kerB, sKer, KER_SLAB);
        __builtin_amdgcn_s_wait_tensorcnt(0);
    }
    __syncthreads();
#endif

#if HAVE_ASYNC_LDS
    for (int i4 = tid; i4 < KER_SLAB / 4; i4 += 128)
        async_copy_b128(kerB + 4*i4, sKer + 4*i4);
    wait_async0();
#else
    for (int i = tid; i < KER_SLAB; i += 128) sKer[i] = kerB[i];
#endif

    // Pull the three group-slice rows we are about to chew on toward L2/L0.
    #pragma unroll
    for (int k = 0; k < 3; ++k) {
        int g = (j + k + 3) & 3;
        __builtin_prefetch(ero + ((((size_t)(b*4 + g)*64 + h)*64 + w)*8)*16, 0, 1);
    }
    __syncthreads();

    float acc[4][16];
    #pragma unroll
    for (int cc = 0; cc < 4; ++cc)
        #pragma unroll
        for (int f = 0; f < 16; ++f) acc[cc][f] = -__builtin_inff();

    for (int k = 0; k < 3; ++k) {
        int g = (j + k + 3) & 3;
        for (int dy = 0; dy < 3; ++dy) {
            int r = h + dy - 1;
            if (r < 0 || r >= 64) continue;     // dilation pad = -inf -> skip
            for (int dx = 0; dx < 3; ++dx) {
                int ww = w + dx - 1;
                if (ww < 0 || ww >= 64) continue;
                const float* ep = ero + ((((size_t)(b*4 + g)*64 + r)*64 + ww)*8 + cbase)*16;
                const float* kp = &sKer[(((dy*3 + dx)*3 + k)*8 + cbase)*16];
                #pragma unroll
                for (int cc = 0; cc < 4; ++cc) {
                    const float4* e4 = reinterpret_cast<const float4*>(ep + cc*16);
                    const float* kq = kp + cc*16;
                    #pragma unroll
                    for (int q = 0; q < 4; ++q) {
                        float4 e = e4[q];
                        acc[cc][4*q+0] = fmaxf(acc[cc][4*q+0], e.x + kq[4*q+0]);
                        acc[cc][4*q+1] = fmaxf(acc[cc][4*q+1], e.y + kq[4*q+1]);
                        acc[cc][4*q+2] = fmaxf(acc[cc][4*q+2], e.z + kq[4*q+2]);
                        acc[cc][4*q+3] = fmaxf(acc[cc][4*q+3], e.w + kq[4*q+3]);
                    }
                }
            }
        }
    }

    // Split each f32 into hi/lo f16 and stage into LDS in (m, c*16+f) order.
    #pragma unroll
    for (int cc = 0; cc < 4; ++cc) {
        int c = cbase + cc;
        #pragma unroll
        for (int p = 0; p < 8; ++p) {
            float v0 = acc[cc][2*p], v1 = acc[cc][2*p + 1];
            _Float16 h0 = (_Float16)v0, h1 = (_Float16)v1;
            _Float16 l0 = (_Float16)(v0 - (float)h0);
            _Float16 l1 = (_Float16)(v1 - (float)h1);
            union { _Float16 hh[2]; unsigned u; } ph, pl;
            ph.hh[0] = h0; ph.hh[1] = h1;
            pl.hh[0] = l0; pl.hh[1] = l1;
            sAcc[0][wv][m][c*8 + p] = ph.u;
            sAcc[1][wv][m][c*8 + p] = pl.u;
        }
    }
    __syncthreads();

    // out[m,n] = sum_k A[m,k] * S[k,n],  S[k,n] = ((k & 15) == n), K = C*F = 128.
    // Per-lane B fragment degenerates to e[j] = (j == lane&15) for both K halves.
    union AF { v16h v; unsigned u[8]; };
    union BF { v16h v; _Float16 e[16]; };
    BF bf;
    #pragma unroll
    for (int jj = 0; jj < 16; ++jj)
        bf.e[jj] = (jj == m) ? (_Float16)1.0f : (_Float16)0.0f;

    v8f d = {};
    #pragma unroll
    for (int plane = 0; plane < 2; ++plane) {
        #pragma unroll
        for (int t = 0; t < 4; ++t) {
            AF af;
            #pragma unroll
            for (int i = 0; i < 8; ++i) {
                // element pair (2i,2i+1) of the A fragment holds K = kg, kg+1
                int kg = 32*t + 16*hi + 2*i;
                af.u[i] = sAcc[plane][wv][m][kg >> 1];
            }
            d = __builtin_amdgcn_wmma_f32_16x16x32_f16(
                    /*neg_a=*/false, af.v, /*neg_b=*/false, bf.v,
                    /*c_mod=*/(short)0, d, /*reuse_a=*/false, /*reuse_b=*/false);
        }
    }

    // D layout: VGPR r, lanes 0-15 -> M=r, lanes 16-31 -> M=8+r; N = lane&15.
    float* ob = out + (((size_t)(b*4 + j)*64 + h)*64)*16;
    #pragma unroll
    for (int r = 0; r < 8; ++r)
        ob[(wv*16 + 8*hi + r)*16 + m] = d[r];
}

extern "C" void kernel_launch(void* const* d_in, const int* in_sizes, int n_in,
                              void* d_out, int out_size, void* d_ws, size_t ws_size,
                              hipStream_t stream) {
    const float* x = (const float*)d_in[0];   // [4,4,64,64,8] f32
    const float* K = (const float*)d_in[1];   // [3,3,3,8,16]  f32
    float* out = (float*)d_out;               // [4,4,64,64,16] f32
    float* ws  = (float*)d_ws;                // needs (32768 + 8388608)*4 B

    prep_kernels<<<(4 * KER_SLAB + 255) / 256, 256, 0, stream>>>(K, ws);
    stageA_erode<<<4 * 4 * 64, 512, 0, stream>>>(x, ws);
    stageB_dilate_wmma<<<4 * 4 * 64, 128, 0, stream>>>(ws, out);
}
